// ViT_71854802862226
// MI455X (gfx1250) — matrix operand order repaired and checked by reading, not compiled, not run on hardware
//
#include <hip/hip_runtime.h>
#include <math.h>

#define NIMG 64
#define SEQ  257
#define DIM  768
#define NH   12
#define DH   64
#define NL   12
#define FF   3072
#define OUTD 10

typedef __attribute__((ext_vector_type(16))) __bf16 v16bf;
typedef __attribute__((ext_vector_type(8)))  float  v8f;

union Frag { v16bf v; unsigned short u[16]; };

__device__ __forceinline__ unsigned short f2bf(float f) {
  unsigned int u = __float_as_uint(f);
  u += 0x7FFFu + ((u >> 16) & 1u);           // round-to-nearest-even
  return (unsigned short)(u >> 16);
}

// CDNA5 async global->LDS copy (ASYNCcnt-tracked, 16B per op)
__device__ __forceinline__ void async_copy_b128(const void* gsrc, void* lds_dst) {
  unsigned lds = (unsigned)(unsigned long long)lds_dst;   // LDS addr = flat[31:0]
  asm volatile("global_load_async_to_lds_b128 %0, %1, off"
               :: "v"(lds), "v"((unsigned long long)gsrc)
               : "memory");
}
__device__ __forceinline__ void wait_async0() {
  asm volatile("s_wait_asynccnt 0x0" ::: "memory");
}

// ---------------------------------------------------------------- f32 -> bf16
__global__ void k_cvt_bf16(const float* __restrict__ src,
                           unsigned short* __restrict__ dst, long n) {
  long i = (long)blockIdx.x * blockDim.x + threadIdx.x;
  if (i < n) dst[i] = f2bf(src[i]);
}

// ---------------------------------------------------------------- patch embed + cls + pos
__global__ void k_embed(const float* __restrict__ img, const float* __restrict__ Wemb,
                        const float* __restrict__ bemb, const float* __restrict__ cls,
                        float* __restrict__ X) {
  long i = (long)blockIdx.x * blockDim.x + threadIdx.x;
  long total = (long)NIMG * SEQ * DIM;
  if (i >= total) return;
  int d = (int)(i % DIM);
  long r = i / DIM;
  int s = (int)(r % SEQ);
  int n = (int)(r / SEQ);
  float v;
  if (s == 0) {
    v = cls[d];
  } else {
    v = bemb[d];
    const float* p = img + (long)n * (256 * 16) + (long)(s - 1) * 16;
    const float* w = Wemb + (long)d * 16;
#pragma unroll
    for (int t = 0; t < 16; ++t) v += p[t] * w[t];
  }
  float inv = __expf(-((float)(d & ~1)) * (9.2103403719761836f / (float)DIM));
  float ang = (float)s * inv;
  v += (d & 1) ? __cosf(ang) : __sinf(ang);
  X[i] = v;
}

// ---------------------------------------------------------------- layernorm -> bf16
__global__ void k_layernorm(const float* __restrict__ X, const float* __restrict__ g,
                            const float* __restrict__ b, unsigned short* __restrict__ H) {
  int wid  = threadIdx.x >> 5;
  int lane = threadIdx.x & 31;
  long row = (long)blockIdx.x * 4 + wid;
  if (row >= (long)NIMG * SEQ) return;
  const float* x = X + row * DIM;
  float vals[24];
  float s = 0.f, ss = 0.f;
#pragma unroll
  for (int j = 0; j < 24; ++j) {
    float v = x[lane + j * 32];
    vals[j] = v; s += v; ss += v * v;
  }
#pragma unroll
  for (int m = 16; m >= 1; m >>= 1) {
    s  += __shfl_xor(s,  m, 32);
    ss += __shfl_xor(ss, m, 32);
  }
  float mu  = s * (1.f / DIM);
  float var = ss * (1.f / DIM) - mu * mu;
  float rs  = rsqrtf(var + 1e-5f);
  unsigned short* h = H + row * DIM;
#pragma unroll
  for (int j = 0; j < 24; ++j) {
    int d = lane + j * 32;
    h[d] = f2bf((vals[j] - mu) * rs * g[d] + b[d]);
  }
}

// ---------------------------------------------------------------- generic bf16 WMMA GEMM
// C[M,N] = A[M,K](bf16 row-major) x Bt[N,K]^T + bias.
// Workgroup = 2 waves = 64x64 macro tile. B panel (64x32/k-step) is staged into
// LDS with async global->LDS copies, double-buffered so the copy of panel i+1
// overlaps the 8 WMMAs on panel i (ASYNCcnt + barrier for handoff).
// MODE 0: store bf16 (QKV) | MODE 1: exact-GELU -> bf16 (fc1) | MODE 2: +residual -> f32 (fc2)
template <int MODE>
__global__ void k_gemm(const unsigned short* __restrict__ A, int lda, int strideAcol,
                       const unsigned short* __restrict__ Bt, int ldb, long strideB,
                       const float* __restrict__ bias, int strideBias,
                       unsigned short* __restrict__ Cb, float* __restrict__ Cf,
                       const float* __restrict__ Cres, int ldc, int strideCcol,
                       int N, int K) {
  __shared__ unsigned short sB[2][64][32];                // 8 KB double buffer
  int ntiles = N / 64;
  int bx = blockIdx.x;
  int mt = bx / ntiles, nt = bx % ntiles;
  int batch = blockIdx.y;
  const unsigned short* Ab = A + batch * strideAcol;
  const unsigned short* Bb = Bt + (long)batch * strideB;
  const float* biasb = bias + batch * strideBias;
  int tid  = threadIdx.x;
  int w    = tid >> 5;
  int lane = tid & 31;
  int half = lane >> 4, lm = lane & 15;
  int row0 = mt * 64 + w * 32;
  int col0 = nt * 64;
  // A element map (contiguous k order): u[e] = arow[k0 + e],
  // +16 per-lane offset for the upper half-wave.
  const unsigned short* arow0 = Ab + (long)(row0 + lm) * lda + (half ? 16 : 0);
  const unsigned short* arow1 = arow0 + (long)16 * lda;
  const unsigned short* bsrc  = Bb + (long)(col0 + tid) * ldb;   // one B row per thread

  v8f acc[2][4] = {};
  int steps = K / 32;
  // stage panel 0
#pragma unroll
  for (int j = 0; j < 4; ++j)
    async_copy_b128(bsrc + j * 8, &sB[0][tid][j * 8]);

  for (int it = 0; it < steps; ++it) {
    int k0  = it * 32;
    int buf = it & 1;
    wait_async0();                    // own async writes to LDS complete
    __syncthreads();                  // all waves' writes visible; prev buffer drained
    if (it + 1 < steps) {             // overlap copy of next panel with compute
      const unsigned short* s = bsrc + k0 + 32;
#pragma unroll
      for (int j = 0; j < 4; ++j)
        async_copy_b128(s + j * 8, &sB[buf ^ 1][tid][j * 8]);
    }
    if (k0 + 64 < K) {                // prefetch next A panel
      __builtin_prefetch(arow0 + k0 + 64, 0, 1);
      __builtin_prefetch(arow1 + k0 + 64, 0, 1);
    }
    Frag a0, a1, bf4[4];
#pragma unroll
    for (int e = 0; e < 8; ++e) {
      a0.u[e]     = arow0[k0 + e];
      a0.u[e + 8] = arow0[k0 + e + 8];
      a1.u[e]     = arow1[k0 + e];
      a1.u[e + 8] = arow1[k0 + e + 8];
    }
#pragma unroll
    for (int o = 0; o < 4; ++o) {
      const unsigned short* bl = &sB[buf][o * 16 + lm][half ? 16 : 0];
#pragma unroll
      for (int e = 0; e < 16; ++e) bf4[o].u[e] = bl[e];
    }
#pragma unroll
    for (int o = 0; o < 4; ++o) {
      acc[0][o] = __builtin_amdgcn_wmma_f32_16x16x32_bf16(false, a0.v, false, bf4[o].v,
                                                          (short)0, acc[0][o], false, false);
      acc[1][o] = __builtin_amdgcn_wmma_f32_16x16x32_bf16(false, a1.v, false, bf4[o].v,
                                                          (short)0, acc[1][o], false, false);
    }
  }
#pragma unroll
  for (int o = 0; o < 4; ++o) {
    float bb = biasb[col0 + o * 16 + lm];
    int c = batch * strideCcol + col0 + o * 16 + lm;
#pragma unroll
    for (int mi = 0; mi < 2; ++mi) {
#pragma unroll
      for (int i = 0; i < 8; ++i) {
        int r = row0 + mi * 16 + i + half * 8;
        long idx = (long)r * ldc + c;
        float v = acc[mi][o][i] + bb;
        if (MODE == 1) v = 0.5f * v * (1.f + erff(v * 0.70710678118654752f));
        if (MODE == 0 || MODE == 1) Cb[idx] = f2bf(v);
        else                        Cf[idx] = Cres[idx] + v;
      }
    }
  }
}

// ---------------------------------------------------------------- attention (flash-lite)
// one wave per (image, head, 16-query tile); scores tile lives in LDS
__global__ void k_attn(const unsigned short* __restrict__ Q,
                       const unsigned short* __restrict__ Kb,
                       const unsigned short* __restrict__ Vb,
                       float* __restrict__ X) {
  __shared__ float          sc[16][288];
  __shared__ unsigned short pb[16][288];
  int bid = blockIdx.x;
  int qt = bid % 17;
  int h  = (bid / 17) % NH;
  int n  = bid / (17 * NH);
  int lane = threadIdx.x & 31;
  int half = lane >> 4, lm = lane & 15;
  const float scale = 0.125f;                       // 1/sqrt(64)
  long base = ((long)n * SEQ) * DIM + h * DH;

  // ---- scores = Q Kt * scale ----
  int sq  = qt * 16 + lm;
  int sqc = sq < SEQ ? sq : SEQ - 1;                // clamp (results discarded)
  const unsigned short* qrow = Q + base + (long)sqc * DIM + (half ? 16 : 0);
  for (int kt = 0; kt < 17; ++kt) {
    int sk  = kt * 16 + lm;
    int skc = sk < SEQ ? sk : SEQ - 1;
    const unsigned short* krow = Kb + base + (long)skc * DIM + (half ? 16 : 0);
    v8f acc = {};
#pragma unroll
    for (int k0 = 0; k0 < DH; k0 += 32) {
      Frag a, b;
#pragma unroll
      for (int e = 0; e < 8; ++e) {
        a.u[e]     = qrow[k0 + e];
        a.u[e + 8] = qrow[k0 + e + 8];
      }
#pragma unroll
      for (int e = 0; e < 16; ++e) b.u[e] = krow[k0 + e];
      acc = __builtin_amdgcn_wmma_f32_16x16x32_bf16(false, a.v, false, b.v,
                                                    (short)0, acc, false, false);
    }
#pragma unroll
    for (int i = 0; i < 8; ++i) {
      int r  = i + half * 8;
      int kg = kt * 16 + lm;
      sc[r][kt * 16 + lm] = (kg < SEQ) ? acc[i] * scale : -3.4e38f;
    }
  }
  __syncthreads();

  // ---- row softmax over 257 keys; P -> bf16 in LDS (zero-padded to 288) ----
  for (int r = 0; r < 16; ++r) {
    float mx = -3.4e38f;
    for (int j = lane; j < SEQ; j += 32) mx = fmaxf(mx, sc[r][j]);
#pragma unroll
    for (int m = 16; m >= 1; m >>= 1) mx = fmaxf(mx, __shfl_xor(mx, m, 32));
    float sum = 0.f;
    for (int j = lane; j < SEQ; j += 32) {
      float e = __expf(sc[r][j] - mx);
      sc[r][j] = e; sum += e;
    }
#pragma unroll
    for (int m = 16; m >= 1; m >>= 1) sum += __shfl_xor(sum, m, 32);
    float inv = 1.f / sum;
    for (int j = lane; j < 288; j += 32)
      pb[r][j] = (j < SEQ) ? f2bf(sc[r][j] * inv) : (unsigned short)0;
  }
  __syncthreads();

  // ---- O = P x V  (K = 288 padded, 4 output tiles of 16) ----
  v8f oacc[4] = {{}, {}, {}, {}};
  for (int kt = 0; kt < 9; ++kt) {
    int k0 = kt * 32;
    Frag a;
#pragma unroll
    for (int e = 0; e < 8; ++e) {
      a.u[e]     = pb[lm][k0 + (half ? 16 : 0) + e];
      a.u[e + 8] = pb[lm][k0 + (half ? 16 : 0) + e + 8];
    }
#pragma unroll
    for (int ot = 0; ot < 4; ++ot) {
      Frag b;
#pragma unroll
      for (int e = 0; e < 16; ++e) {
        int kg = k0 + (half ? 16 : 0) + e;
        unsigned short v = 0;
        if (kg < SEQ) v = Vb[base + (long)kg * DIM + ot * 16 + lm];
        b.u[e] = v;
      }
      oacc[ot] = __builtin_amdgcn_wmma_f32_16x16x32_bf16(false, a.v, false, b.v,
                                                         (short)0, oacc[ot], false, false);
    }
  }
  // residual add fused: X += O
#pragma unroll
  for (int ot = 0; ot < 4; ++ot) {
#pragma unroll
    for (int i = 0; i < 8; ++i) {
      int r = i + half * 8;
      int s = qt * 16 + r;
      if (s < SEQ) {
        long idx = base + (long)s * DIM + ot * 16 + lm;
        X[idx] += oacc[ot][i];
      }
    }
  }
}

// ---------------------------------------------------------------- classifier head + softmax
__global__ void k_head(const float* __restrict__ X, const float* __restrict__ Wh,
                       const float* __restrict__ bh, float* __restrict__ out) {
  __shared__ float lg[OUTD];
  int n = blockIdx.x, t = threadIdx.x;
  const float* x = X + (long)n * SEQ * DIM;           // token 0
  if (t < OUTD) {
    float s = bh[t];
    const float* w = Wh + (long)t * DIM;
    for (int d = 0; d < DIM; ++d) s += x[d] * w[d];
    lg[t] = s;
  }
  __syncthreads();
  if (t == 0) {
    float mx = lg[0];
    for (int j = 1; j < OUTD; ++j) mx = fmaxf(mx, lg[j]);
    float sum = 0.f;
    for (int j = 0; j < OUTD; ++j) { float e = __expf(lg[j] - mx); lg[j] = e; sum += e; }
    float inv = 1.f / sum;
    for (int j = 0; j < OUTD; ++j) out[n * OUTD + j] = lg[j] * inv;
  }
}

// ================================================================ launch
extern "C" void kernel_launch(void* const* d_in, const int* in_sizes, int n_in,
                              void* d_out, int out_size, void* d_ws, size_t ws_size,
                              hipStream_t stream) {
  const float* images = (const float*)d_in[0];
  const float* Wemb   = (const float*)d_in[1];
  const float* bemb   = (const float*)d_in[2];
  const float* cls    = (const float*)d_in[3];
  const float* Wq     = (const float*)d_in[4];
  const float* bq     = (const float*)d_in[5];
  const float* Wk     = (const float*)d_in[6];
  const float* bk     = (const float*)d_in[7];
  const float* Wv     = (const float*)d_in[8];
  const float* bv     = (const float*)d_in[9];
  const float* ln1g   = (const float*)d_in[10];
  const float* ln1b   = (const float*)d_in[11];
  const float* ln2g   = (const float*)d_in[12];
  const float* ln2b   = (const float*)d_in[13];
  const float* W1     = (const float*)d_in[14];
  const float* b1     = (const float*)d_in[15];
  const float* W2     = (const float*)d_in[16];
  const float* b2     = (const float*)d_in[17];
  const float* Wh     = (const float*)d_in[18];
  const float* bh     = (const float*)d_in[19];

  char* ws = (char*)d_ws;
  size_t off = 0;
  auto alloc = [&](size_t bytes) {
    void* p = ws + off;
    off += (bytes + 255) & ~(size_t)255;
    return p;
  };
  const long NSD = (long)NIMG * SEQ * DIM;            // 12,632,064
  float*          X    = (float*)alloc(NSD * 4);
  unsigned short* Hb   = (unsigned short*)alloc(NSD * 2);
  unsigned short* Qb   = (unsigned short*)alloc(NSD * 2);
  unsigned short* Kbuf = (unsigned short*)alloc(NSD * 2);
  unsigned short* Vbuf = (unsigned short*)alloc(NSD * 2);
  unsigned short* Mb   = (unsigned short*)alloc((long)NIMG * SEQ * FF * 2);
  const long WQKV = (long)NL * NH * DH * DH;
  unsigned short* Wqb = (unsigned short*)alloc(WQKV * 2);
  unsigned short* Wkb = (unsigned short*)alloc(WQKV * 2);
  unsigned short* Wvb = (unsigned short*)alloc(WQKV * 2);
  const long WFF = (long)NL * FF * DIM;
  unsigned short* W1b = (unsigned short*)alloc(WFF * 2);
  unsigned short* W2b = (unsigned short*)alloc(WFF * 2);

  // one-time weight conversion to bf16
  k_cvt_bf16<<<(int)((WQKV + 255) / 256), 256, 0, stream>>>(Wq, Wqb, WQKV);
  k_cvt_bf16<<<(int)((WQKV + 255) / 256), 256, 0, stream>>>(Wk, Wkb, WQKV);
  k_cvt_bf16<<<(int)((WQKV + 255) / 256), 256, 0, stream>>>(Wv, Wvb, WQKV);
  k_cvt_bf16<<<(int)((WFF + 255) / 256), 256, 0, stream>>>(W1, W1b, WFF);
  k_cvt_bf16<<<(int)((WFF + 255) / 256), 256, 0, stream>>>(W2, W2b, WFF);

  k_embed<<<(int)((NSD + 255) / 256), 256, 0, stream>>>(images, Wemb, bemb, cls, X);

  const int ROWS = NIMG * SEQ;                        // 16448 = 257 * 64
  for (int l = 0; l < NL; ++l) {
    k_layernorm<<<(ROWS + 3) / 4, 128, 0, stream>>>(X, ln1g + l * DIM, ln1b + l * DIM, Hb);

    dim3 gq(ROWS / 64, NH);                           // N=64 -> one 64-col tile
    k_gemm<0><<<gq, 64, 0, stream>>>(Hb, DIM, DH, Wqb + (long)l * NH * DH * DH, DH,
                                     (long)DH * DH, bq + l * NH * DH, DH,
                                     Qb, nullptr, nullptr, DIM, DH, DH, DH);
    k_gemm<0><<<gq, 64, 0, stream>>>(Hb, DIM, DH, Wkb + (long)l * NH * DH * DH, DH,
                                     (long)DH * DH, bk + l * NH * DH, DH,
                                     Kbuf, nullptr, nullptr, DIM, DH, DH, DH);
    k_gemm<0><<<gq, 64, 0, stream>>>(Hb, DIM, DH, Wvb + (long)l * NH * DH * DH, DH,
                                     (long)DH * DH, bv + l * NH * DH, DH,
                                     Vbuf, nullptr, nullptr, DIM, DH, DH, DH);

    k_attn<<<NIMG * NH * 17, 32, 0, stream>>>(Qb, Kbuf, Vbuf, X);

    k_layernorm<<<(ROWS + 3) / 4, 128, 0, stream>>>(X, ln2g + l * DIM, ln2b + l * DIM, Hb);

    dim3 g1((ROWS / 64) * (FF / 64), 1);
    k_gemm<1><<<g1, 64, 0, stream>>>(Hb, DIM, 0, W1b + (long)l * FF * DIM, DIM, 0,
                                     b1 + l * FF, 0, Mb, nullptr, nullptr, FF, 0, FF, DIM);
    dim3 g2((ROWS / 64) * (DIM / 64), 1);
    k_gemm<2><<<g2, 64, 0, stream>>>(Mb, FF, 0, W2b + (long)l * DIM * FF, FF, 0,
                                     b2 + l * DIM, 0, nullptr, X, X, DIM, 0, DIM, FF);
  }

  k_head<<<NIMG, 32, 0, stream>>>(X, Wh, bh, (float*)d_out);
  (void)in_sizes; (void)n_in; (void)out_size; (void)ws_size;
}
